// NolinearAttention_35253091565648
// MI455X (gfx1250) — matrix operand order repaired and checked by reading, not compiled, not run on hardware
//
#include <hip/hip_runtime.h>
#include <hip/hip_bf16.h>
#include <math.h>

typedef __bf16 bf16_t;
typedef bf16_t bf16x4  __attribute__((ext_vector_type(4)));
typedef bf16_t bf16x8  __attribute__((ext_vector_type(8)));
typedef bf16_t bf16x16 __attribute__((ext_vector_type(16)));
typedef float  v8f     __attribute__((ext_vector_type(8)));
typedef int    v4i     __attribute__((vector_size(16)));

#define B_    32
#define S_    2048
#define HE_   1024
#define HD_   1024
#define WSTR  2048                      // HE + HD, row stride of W
#define KSLAB 128
#define NSLAB (HE_ / KSLAB)             // 8
#define F4T   ((32 * KSLAB / 4) / 256)  // float4 copies per thread per slab = 4

#define AS1 __attribute__((address_space(1)))
#define AS3 __attribute__((address_space(3)))

#if defined(__has_builtin)
#if __has_builtin(__builtin_amdgcn_global_load_async_to_lds_b128)
#define HAVE_ASYNC_LDS 1
#endif
#endif

template <int N>
static __device__ __forceinline__ void wait_asynccnt() {
#if defined(__has_builtin) && __has_builtin(__builtin_amdgcn_s_wait_asynccnt)
    __builtin_amdgcn_s_wait_asynccnt(N);
#else
    asm volatile("s_wait_asynccnt %0" ::"i"(N) : "memory");
#endif
}

#if defined(HAVE_ASYNC_LDS)
static __device__ __forceinline__ void async_copy_b128(const float* g, float* l) {
    __builtin_amdgcn_global_load_async_to_lds_b128((AS1 v4i*)g, (AS3 v4i*)l, 0, 0);
}
#endif

static __device__ __forceinline__ v8f wmma_bf16(bf16x16 a, bf16x16 b, v8f c) {
    return __builtin_amdgcn_wmma_f32_16x16x32_bf16(false, a, false, b, (short)0, c,
                                                   false, false);
}

// ---------------- kernel 1: W_enc (f32, strided) -> Wb (bf16, dense [d][e]) --------
__global__ __launch_bounds__(256) void k_prep_wenc(const float* __restrict__ W,
                                                   bf16_t* __restrict__ Wb) {
    int idx = blockIdx.x * 256 + threadIdx.x;   // d*1024 + e
    int d = idx >> 10, e = idx & 1023;
    Wb[idx] = (bf16_t)W[(size_t)d * WSTR + e];
}

// ---------------- kernel 2: proj[b][d] = dec[b] . W_dec[d] (f32 VALU, tiny) --------
__global__ __launch_bounds__(256) void k_proj(const float* __restrict__ dec,
                                              const float* __restrict__ W,
                                              float* __restrict__ proj) {
    int idx = blockIdx.x * 256 + threadIdx.x;   // b*1024 + d
    int b = idx >> 10, d = idx & 1023;
    const float4* wr = (const float4*)(W + (size_t)d * WSTR + HE_);
    const float4* dr = (const float4*)(dec + (size_t)b * HD_);
    float acc = 0.f;
    #pragma unroll 4
    for (int q = 0; q < HD_ / 4; ++q) {
        float4 w = wr[q], x = dr[q];
        acc += w.x * x.x + w.y * x.y + w.z * x.z + w.w * x.w;
    }
    proj[idx] = acc;
}

// ---------------- kernel 3: fused  scores = V . tanh(enc@W_enc^T + proj) -----------
// One WG = 32 (b,s)-rows x all 1024 d.  8 waves, each: 2 M-tiles x 8 d-tiles.
// A slabs staged via async DMA to LDS (double-buffered f32) then converted to bf16.
__global__ __launch_bounds__(256) void k_scores(const float* __restrict__ enc,
                                                const bf16_t* __restrict__ Wb,
                                                const float* __restrict__ proj,
                                                const float* __restrict__ Vw,
                                                float* __restrict__ scores) {
#if defined(HAVE_ASYNC_LDS)
    __shared__ __align__(16) float ldsF[2][32 * KSLAB];   // 32 KB ping-pong
#endif
    __shared__ __align__(16) bf16_t ldsA[32 * KSLAB];     // 8 KB bf16 A slab
    __shared__ float ldsScore[32];

    const int tid  = threadIdx.x;
    const int wave = tid >> 5;
    const int lane = tid & 31;
    const int ln15 = lane & 15;
    const int hi   = lane >> 4;            // 0: lanes 0-15, 1: lanes 16-31
    const int rowBase = blockIdx.x * 32;   // flat (b*S + s) row
    const int bIdx = rowBase / S_;

    if (tid < 32) ldsScore[tid] = 0.f;

    v8f acc[2][8];
    const v8f vzero = {0.f, 0.f, 0.f, 0.f, 0.f, 0.f, 0.f, 0.f};
    #pragma unroll
    for (int t = 0; t < 2; ++t)
        #pragma unroll
        for (int j = 0; j < 8; ++j) acc[t][j] = vzero;

#if defined(HAVE_ASYNC_LDS)
    // prime the pipeline: DMA slab 0 into buffer 0
    #pragma unroll
    for (int i = 0; i < F4T; ++i) {
        int lin = tid + i * 256;                  // float4 index within slab
        int row = lin >> 5, c = (lin & 31) << 2;  // 32 float4 per row
        async_copy_b128(&enc[(size_t)(rowBase + row) * HE_ + c],
                        &ldsF[0][row * KSLAB + c]);
    }
#endif

    for (int slab = 0; slab < NSLAB; ++slab) {
#if defined(HAVE_ASYNC_LDS)
        const int cur = slab & 1;
        if (slab + 1 < NSLAB) {
            #pragma unroll
            for (int i = 0; i < F4T; ++i) {       // DMA next slab while we compute
                int lin = tid + i * 256;
                int row = lin >> 5, c = (lin & 31) << 2;
                async_copy_b128(
                    &enc[(size_t)(rowBase + row) * HE_ + (slab + 1) * KSLAB + c],
                    &ldsF[1 - cur][row * KSLAB + c]);
            }
            wait_asynccnt<F4T>();                 // current slab landed (in-order)
        } else {
            wait_asynccnt<0>();
        }
        __syncthreads();
        // convert f32 slab -> bf16 fragment buffer
        #pragma unroll
        for (int i = 0; i < F4T; ++i) {
            int q = (tid + i * 256) << 2;
            float4 v = *(const float4*)&ldsF[cur][q];
            bf16x4 h;
            h[0] = (bf16_t)v.x; h[1] = (bf16_t)v.y;
            h[2] = (bf16_t)v.z; h[3] = (bf16_t)v.w;
            *(bf16x4*)&ldsA[q] = h;
        }
        __syncthreads();
#else
        __syncthreads();
        #pragma unroll
        for (int i = 0; i < F4T; ++i) {           // synchronous staging fallback
            int lin = tid + i * 256;
            int row = lin >> 5, c = (lin & 31) << 2;
            float4 v = *(const float4*)&enc[(size_t)(rowBase + row) * HE_ +
                                            slab * KSLAB + c];
            bf16x4 h;
            h[0] = (bf16_t)v.x; h[1] = (bf16_t)v.y;
            h[2] = (bf16_t)v.z; h[3] = (bf16_t)v.w;
            *(bf16x4*)&ldsA[row * KSLAB + c] = h;
        }
        __syncthreads();
#endif
        if (slab + 1 < NSLAB)                     // warm L2/WGP$ for next W slab
            __builtin_prefetch(Wb + (size_t)((wave * 8) * 16 + ln15) * HE_ +
                                   (slab + 1) * KSLAB + hi * 16, 0, 0);

        for (int kt = 0; kt < KSLAB / 32; ++kt) {
            // A frags (CDNA5: row = lane&15; elems 0-7 K=hi*8.., 8-15 K=16+hi*8..)
            const int klo = kt * 32 + hi * 8;
            bf16x8 a0l = *(const bf16x8*)&ldsA[ln15 * KSLAB + klo];
            bf16x8 a0h = *(const bf16x8*)&ldsA[ln15 * KSLAB + klo + 16];
            bf16x16 a0 = __builtin_shufflevector(a0l, a0h, 0,1,2,3,4,5,6,7,
                                                 8,9,10,11,12,13,14,15);
            bf16x8 a1l = *(const bf16x8*)&ldsA[(16 + ln15) * KSLAB + klo];
            bf16x8 a1h = *(const bf16x8*)&ldsA[(16 + ln15) * KSLAB + klo + 16];
            bf16x16 a1 = __builtin_shufflevector(a1l, a1h, 0,1,2,3,4,5,6,7,
                                                 8,9,10,11,12,13,14,15);

            auto load_b = [&](int j) -> bf16x16 {
                // B frag (lane = col; elems 0-7 K=hi*8.., 8-15 K=16+hi*8..)
                int d = (wave * 8 + j) * 16 + ln15;
                const bf16_t* bp = Wb + (size_t)d * HE_ + slab * KSLAB +
                                   kt * 32 + hi * 8;
                bf16x8 bl = *(const bf16x8*)bp;
                bf16x8 bh = *(const bf16x8*)(bp + 16);
                return __builtin_shufflevector(bl, bh, 0,1,2,3,4,5,6,7,
                                               8,9,10,11,12,13,14,15);
            };
            bf16x16 bcur = load_b(0);
            #pragma unroll
            for (int j = 0; j < 8; ++j) {         // keep next B in flight over WMMAs
                bf16x16 bnxt = bcur;
                if (j < 7) bnxt = load_b(j + 1);
                acc[0][j] = wmma_bf16(a0, bcur, acc[0][j]);
                acc[1][j] = wmma_bf16(a1, bcur, acc[1][j]);
                bcur = bnxt;
            }
        }
    }

    // epilogue: tanh + V-dot, per-lane partials for its 8 rows (m = v + hi*8)
    float sp[2][8];
    #pragma unroll
    for (int t = 0; t < 2; ++t)
        #pragma unroll
        for (int v = 0; v < 8; ++v) sp[t][v] = 0.f;

    #pragma unroll
    for (int t = 0; t < 2; ++t) {
        #pragma unroll
        for (int j = 0; j < 8; ++j) {
            int dg = (wave * 8 + j) * 16 + ln15;
            float pv = proj[(size_t)bIdx * HD_ + dg];
            float vw = Vw[dg];
            #pragma unroll
            for (int v = 0; v < 8; ++v)
                sp[t][v] += tanhf(acc[t][j][v] + pv) * vw;
        }
    }
    #pragma unroll
    for (int t = 0; t < 2; ++t)
        #pragma unroll
        for (int v = 0; v < 8; ++v) {
            float s = sp[t][v];
            s += __shfl_xor(s, 1);
            s += __shfl_xor(s, 2);
            s += __shfl_xor(s, 4);
            s += __shfl_xor(s, 8);
            sp[t][v] = s;
        }
    if (lane == 0 || lane == 16) {
        #pragma unroll
        for (int t = 0; t < 2; ++t)
            #pragma unroll
            for (int v = 0; v < 8; ++v)
                atomicAdd(&ldsScore[t * 16 + hi * 8 + v], sp[t][v]);
    }
    __syncthreads();
    if (tid < 32)
        scores[(size_t)bIdx * S_ + (rowBase % S_) + tid] = ldsScore[tid];
}

// ---------------- kernel 4: masked softmax over S per batch ------------------------
__global__ __launch_bounds__(256) void k_softmax(const float* __restrict__ scores,
                                                 const unsigned char* __restrict__ mask,
                                                 float* __restrict__ attn) {
    __shared__ float red[256];
    const int b = blockIdx.x, tid = threadIdx.x;
    float xs[S_ / 256];
    float mx = -INFINITY;
    #pragma unroll
    for (int i = 0; i < S_ / 256; ++i) {
        int s = tid + i * 256;
        float x = scores[(size_t)b * S_ + s];
        if (mask[(size_t)b * S_ + s]) x = -INFINITY;
        xs[i] = x;
        mx = fmaxf(mx, x);
    }
    red[tid] = mx; __syncthreads();
    for (int o = 128; o > 0; o >>= 1) {
        if (tid < o) red[tid] = fmaxf(red[tid], red[tid + o]);
        __syncthreads();
    }
    mx = red[0]; __syncthreads();
    float sum = 0.f;
    #pragma unroll
    for (int i = 0; i < S_ / 256; ++i) {
        xs[i] = __expf(xs[i] - mx);
        sum += xs[i];
    }
    red[tid] = sum; __syncthreads();
    for (int o = 128; o > 0; o >>= 1) {
        if (tid < o) red[tid] += red[tid + o];
        __syncthreads();
    }
    float inv = 1.f / red[0];
    #pragma unroll
    for (int i = 0; i < S_ / 256; ++i)
        attn[(size_t)b * S_ + tid + i * 256] = xs[i] * inv;
}

// ---------------- kernel 5: ctx[b][e] = sum_s attn[b][s] * enc[b][s][e] ------------
__global__ __launch_bounds__(256) void k_ctx(const float* __restrict__ enc,
                                             const float* __restrict__ attn,
                                             float* __restrict__ ctx) {
    __shared__ float sa[S_];
    const int b = blockIdx.x >> 2;
    const int eBase = (blockIdx.x & 3) * 256;
    const int tid = threadIdx.x;
    for (int i = tid; i < S_; i += 256) sa[i] = attn[(size_t)b * S_ + i];
    __syncthreads();
    float acc = 0.f;
    const float* ep = enc + (size_t)b * S_ * HE_ + eBase + tid;
    #pragma unroll 4
    for (int s = 0; s < S_; ++s) acc += sa[s] * ep[(size_t)s * HE_];
    ctx[(size_t)b * HE_ + eBase + tid] = acc;
}

extern "C" void kernel_launch(void* const* d_in, const int* in_sizes, int n_in,
                              void* d_out, int out_size, void* d_ws, size_t ws_size,
                              hipStream_t stream) {
    const float*         enc  = (const float*)d_in[0];          // (B,S,HE)
    const float*         dec  = (const float*)d_in[1];          // (B,HD)
    const unsigned char* mask = (const unsigned char*)d_in[2];  // (B,S) bool
    const float*         W    = (const float*)d_in[3];          // (HD, HE+HD)
    const float*         V    = (const float*)d_in[4];          // (1, HD)

    float* out  = (float*)d_out;
    float* ctx  = out;               // B*HE floats
    float* attn = out + B_ * HE_;    // B*S floats

    char*   ws     = (char*)d_ws;
    bf16_t* Wb     = (bf16_t*)ws;                                       // 2 MB
    float*  proj   = (float*)(ws + (size_t)HD_ * HE_ * sizeof(bf16_t)); // 128 KB
    float*  scores = (float*)((char*)proj + (size_t)B_ * HD_ * sizeof(float));

    k_prep_wenc<<<HD_ * HE_ / 256, 256, 0, stream>>>(W, Wb);
    k_proj     <<<B_ * HD_ / 256, 256, 0, stream>>>(dec, W, proj);
    k_scores   <<<(B_ * S_) / 32, 256, 0, stream>>>(enc, Wb, proj, V, scores);
    k_softmax  <<<B_, 256, 0, stream>>>(scores, mask, attn);
    k_ctx      <<<B_ * (HE_ / 256), 256, 0, stream>>>(enc, attn, ctx);
}
